// HGRNBitMLP_42064909697513
// MI455X (gfx1250) — 8-bit matrix operand order repaired and checked by reading, compiled, not run on hardware
//
#include <hip/hip_runtime.h>
#include <hip/hip_bf16.h>
#include <stdint.h>


typedef __attribute__((ext_vector_type(8))) int v8i;

#define HDIM 2048
#define IDIM 8192
#define MTOK 4096
#define EPS_NORM 1e-8f
#define EPS_Q    1e-5f


__device__ __forceinline__ v8i load_a8(const int8_t* p) {
    const int2 t0 = *(const int2*)(p +  0);
    const int2 t1 = *(const int2*)(p +  8);
    const int2 t2 = *(const int2*)(p + 32);
    const int2 t3 = *(const int2*)(p + 40);
    v8i a;
    a[0] = t0.x; a[1] = t0.y; a[2] = t1.x; a[3] = t1.y;
    a[4] = t2.x; a[5] = t2.y; a[6] = t3.x; a[7] = t3.y;
    return a;
}

__device__ __forceinline__ v8i load_b8(const int8_t* p) {
    const int4 t0 = *(const int4*)(p +  0);
    const int4 t1 = *(const int4*)(p + 32);
    v8i b;
    b[0] = t0.x; b[1] = t0.y; b[2] = t0.z; b[3] = t0.w;
    b[4] = t1.x; b[5] = t1.y; b[6] = t1.z; b[7] = t1.w;
    return b;
}

__device__ __forceinline__ v8i wmma_iu8(v8i a, v8i b, v8i c) {
    return __builtin_amdgcn_wmma_i32_16x16x64_iu8(true, a, true, b, c, false, false);
}


__global__ void __launch_bounds__(256)
absmean_partial_kernel(const float* __restrict__ w, int n, float* __restrict__ partial) {
    __shared__ float red[256];
    const int tid = threadIdx.x;
    float s = 0.0f;
    for (int i = blockIdx.x * 256 + tid; i < n; i += gridDim.x * 256)
        s += fabsf(w[i]);
    red[tid] = s;
    __syncthreads();
    for (int st = 128; st > 0; st >>= 1) {
        if (tid < st) red[tid] += red[tid + st];
        __syncthreads();
    }
    if (tid == 0) partial[blockIdx.x] = red[0];
}

__global__ void __launch_bounds__(256)
absmean_final_kernel(const float* __restrict__ partial, int nparts, float inv_count,
                     float* __restrict__ wdeq_out) {
    __shared__ float red[256];
    const int tid = threadIdx.x;
    float s = 0.0f;
    for (int i = tid; i < nparts; i += 256) s += partial[i];
    red[tid] = s;
    __syncthreads();
    for (int st = 128; st > 0; st >>= 1) {
        if (tid < st) red[tid] += red[tid + st];
        __syncthreads();
    }
    if (tid == 0) wdeq_out[0] = fmaxf(red[0] * inv_count, EPS_Q);
}


__global__ void __launch_bounds__(256)
wquant_kernel(const float* __restrict__ w, const float* __restrict__ wdeqp,
              int8_t* __restrict__ out, int n) {
    const float scale = 1.0f / wdeqp[0];
    const int stride = gridDim.x * blockDim.x;
    for (int i = blockIdx.x * blockDim.x + threadIdx.x; i < n; i += stride) {
        float q = rintf(w[i] * scale);
        q = fminf(fmaxf(q, -1.0f), 1.0f);
        out[i] = (int8_t)(int)q;
    }
}


template<int L, int NT>
__global__ void __launch_bounds__(NT)
actquant_kernel(const float* __restrict__ x, const float* __restrict__ g,
                int8_t* __restrict__ aq, float* __restrict__ adeq) {
    __shared__ float pbuf[L];
    __shared__ float rss[NT];
    __shared__ float rmx[NT];
    const int row = blockIdx.x;
    const int tid = threadIdx.x;
    const float* xr = x + (size_t)row * L;
    float ss = 0.0f, mx = 0.0f;
    for (int i = tid; i < L; i += NT) {
        const float xv = xr[i];
        const float p  = xv * g[i];
        pbuf[i] = p;
        ss += xv * xv;
        mx = fmaxf(mx, fabsf(p));
    }
    rss[tid] = ss; rmx[tid] = mx;
    __syncthreads();
    for (int st = NT / 2; st > 0; st >>= 1) {
        if (tid < st) {
            rss[tid] += rss[tid + st];
            rmx[tid] = fmaxf(rmx[tid], rmx[tid + st]);
        }
        __syncthreads();
    }
    const float rstd = rsqrtf(rss[0] / (float)L + EPS_NORM);
    const float amax = fmaxf(rmx[0] * rstd, EPS_Q);
    const float scl  = 127.0f / amax;
    if (tid == 0) adeq[row] = amax * (1.0f / 127.0f);
    int8_t* ar = aq + (size_t)row * L;
    for (int i = tid; i < L; i += NT) {
        float q = rintf(pbuf[i] * rstd * scl);
        q = fminf(fmaxf(q, -128.0f), 127.0f);
        ar[i] = (int8_t)(int)q;
    }
}


__global__ void __launch_bounds__(256)
gemm1_gate_kernel(const int8_t* __restrict__ aq, const float* __restrict__ adeq,
                  const int8_t* __restrict__ wq, const float* __restrict__ wdeqp,
                  float* __restrict__ z) {
    const int tid  = threadIdx.x;
    const int wave = tid >> 5;
    const int lane = tid & 31;
    const int half = lane >> 4;
    const int r16  = lane & 15;
    const int mrow = blockIdx.y * 16;
    const int nbase = blockIdx.x * 256 + wave * 32;

    const int8_t* ap  = aq + (size_t)(mrow + r16) * HDIM + half * 16;
    const int8_t* bg0 = wq + (size_t)(nbase        + r16) * HDIM + half * 16;
    const int8_t* bg1 = wq + (size_t)(nbase + 16   + r16) * HDIM + half * 16;
    const int8_t* bu0 = wq + (size_t)(nbase + IDIM + r16) * HDIM + half * 16;
    const int8_t* bu1 = wq + (size_t)(nbase + IDIM + 16 + r16) * HDIM + half * 16;

    v8i acc_g0 = {0,0,0,0,0,0,0,0};
    v8i acc_g1 = {0,0,0,0,0,0,0,0};
    v8i acc_u0 = {0,0,0,0,0,0,0,0};
    v8i acc_u1 = {0,0,0,0,0,0,0,0};

    for (int k = 0; k < HDIM; k += 64) {
        const v8i a = load_a8(ap + k);
        acc_g0 = wmma_iu8(a, load_b8(bg0 + k), acc_g0);
        acc_g1 = wmma_iu8(a, load_b8(bg1 + k), acc_g1);
        acc_u0 = wmma_iu8(a, load_b8(bu0 + k), acc_u0);
        acc_u1 = wmma_iu8(a, load_b8(bu1 + k), acc_u1);
    }

    const float wdeq = wdeqp[0];
    #pragma unroll
    for (int r = 0; r < 8; ++r) {
        const int row = mrow + half * 8 + r;
        const float d = adeq[row] * wdeq;
        const float g0 = (float)acc_g0[r] * d;
        const float u0 = (float)acc_u0[r] * d;
        const float g1 = (float)acc_g1[r] * d;
        const float u1 = (float)acc_u1[r] * d;
        const float s0 = g0 / (1.0f + expf(-g0));
        const float s1 = g1 / (1.0f + expf(-g1));
        float* zr = z + (size_t)row * IDIM;
        zr[nbase + r16]      = s0 * u0;
        zr[nbase + 16 + r16] = s1 * u1;
    }
}


__global__ void __launch_bounds__(256)
gemm2_down_kernel(const int8_t* __restrict__ aq, const float* __restrict__ adeq,
                  const int8_t* __restrict__ wq, const float* __restrict__ wdeqp,
                  float* __restrict__ out) {
    const int tid  = threadIdx.x;
    const int wave = tid >> 5;
    const int lane = tid & 31;
    const int half = lane >> 4;
    const int r16  = lane & 15;
    const int mrow = blockIdx.y * 16;
    const int nbase = blockIdx.x * 512 + wave * 64;

    const int8_t* ap = aq + (size_t)(mrow + r16) * IDIM + half * 16;
    const int8_t* b0 = wq + (size_t)(nbase      + r16) * IDIM + half * 16;
    const int8_t* b1 = wq + (size_t)(nbase + 16 + r16) * IDIM + half * 16;
    const int8_t* b2 = wq + (size_t)(nbase + 32 + r16) * IDIM + half * 16;
    const int8_t* b3 = wq + (size_t)(nbase + 48 + r16) * IDIM + half * 16;

    v8i a0 = {0,0,0,0,0,0,0,0};
    v8i a1 = {0,0,0,0,0,0,0,0};
    v8i a2 = {0,0,0,0,0,0,0,0};
    v8i a3 = {0,0,0,0,0,0,0,0};

    for (int k = 0; k < IDIM; k += 64) {
        const v8i a = load_a8(ap + k);
        a0 = wmma_iu8(a, load_b8(b0 + k), a0);
        a1 = wmma_iu8(a, load_b8(b1 + k), a1);
        a2 = wmma_iu8(a, load_b8(b2 + k), a2);
        a3 = wmma_iu8(a, load_b8(b3 + k), a3);
    }

    const float wdeq = wdeqp[0];
    #pragma unroll
    for (int r = 0; r < 8; ++r) {
        const int row = mrow + half * 8 + r;
        const float d = adeq[row] * wdeq;
        float* orow = out + (size_t)row * HDIM;
        orow[nbase      + r16] = (float)a0[r] * d;
        orow[nbase + 16 + r16] = (float)a1[r] * d;
        orow[nbase + 32 + r16] = (float)a2[r] * d;
        orow[nbase + 48 + r16] = (float)a3[r] * d;
    }
}


extern "C" void kernel_launch(void* const* d_in, const int* in_sizes, int n_in,
                              void* d_out, int out_size, void* d_ws, size_t ws_size,
                              hipStream_t stream) {
    const float* x      = (const float*)d_in[0];
    const float* w_gate = (const float*)d_in[1];
    const float* g_gate = (const float*)d_in[2];
    const float* w_down = (const float*)d_in[3];
    const float* g_down = (const float*)d_in[4];
    float* out = (float*)d_out;

    const int n_wgate = 2 * IDIM * HDIM;
    const int n_wdown = HDIM * IDIM;

    char* ws = (char*)d_ws;
    float*  wdeq_gate = (float*)(ws + 0);
    float*  wdeq_down = (float*)(ws + 256);
    float*  partials  = (float*)(ws + 512);
    int8_t* wq_gate   = (int8_t*)(ws + 8192);
    int8_t* wq_down   = wq_gate + (size_t)n_wgate;
    int8_t* aq1       = wq_down + (size_t)n_wdown;
    int8_t* aq2       = aq1 + (size_t)MTOK * HDIM;
    float*  adeq1     = (float*)(aq2 + (size_t)MTOK * IDIM);
    float*  adeq2     = adeq1 + MTOK;
    float*  zbuf      = (float*)(adeq2 + MTOK);

    absmean_partial_kernel<<<1024, 256, 0, stream>>>(w_gate, n_wgate, partials);
    absmean_final_kernel<<<1, 256, 0, stream>>>(partials, 1024, 1.0f / (float)n_wgate, wdeq_gate);
    absmean_partial_kernel<<<1024, 256, 0, stream>>>(w_down, n_wdown, partials);
    absmean_final_kernel<<<1, 256, 0, stream>>>(partials, 1024, 1.0f / (float)n_wdown, wdeq_down);

    wquant_kernel<<<8192, 256, 0, stream>>>(w_gate, wdeq_gate, wq_gate, n_wgate);
    wquant_kernel<<<8192, 256, 0, stream>>>(w_down, wdeq_down, wq_down, n_wdown);

    actquant_kernel<HDIM, 256><<<MTOK, 256, 0, stream>>>(x, g_gate, aq1, adeq1);

    gemm1_gate_kernel<<<dim3(IDIM / 256, MTOK / 16), 256, 0, stream>>>(
        aq1, adeq1, wq_gate, wdeq_gate, zbuf);

    actquant_kernel<IDIM, 256><<<MTOK, 256, 0, stream>>>(zbuf, g_down, aq2, adeq2);

    gemm2_down_kernel<<<dim3(HDIM / 512, MTOK / 16), 256, 0, stream>>>(
        aq2, adeq2, wq_down, wdeq_down, out);
}
